// Attention_2018634629856
// MI455X (gfx1250) — matrix operand order repaired and checked by reading, not compiled, not run on hardware
//
#include <hip/hip_runtime.h>
#include <hip/hip_bf16.h>

#define DIM   2048
#define NH    16
#define HD    128
#define BATCH 32
#define SEQ   256
#define MROWS (BATCH * SEQ)   // 8192

typedef __bf16 bf16_t;
typedef __attribute__((ext_vector_type(16))) __bf16 v16bf;
typedef __attribute__((ext_vector_type(8)))  __bf16 v8bf;
typedef __attribute__((ext_vector_type(8)))  float  v8f;

__device__ __forceinline__ v8f wmma_bf16(v16bf a, v16bf b, v8f c) {
  // D = A(16x32 bf16) * B(32x16 bf16) + C(16x16 f32)
  return __builtin_amdgcn_wmma_f32_16x16x32_bf16(
      /*neg_a=*/false, a, /*neg_b=*/false, b,
      /*c_mod=*/(short)0, c, /*reuse_a=*/false, /*reuse_b=*/false);
}

// A-fragment (16x32, M x K): lane L<16 -> row M=L, K 0..15;
// lane L>=16 -> row M=L-16, K 16..31. `row32` points at this
// lane's 32-element K-slice (bf16, 16B-aligned chunks).
__device__ __forceinline__ v16bf ld_frag_a(const bf16_t* row32, int lane) {
  const int k0 = (lane < 16) ? 0 : 16;
  v8bf lo = *reinterpret_cast<const v8bf*>(row32 + k0);
  v8bf hi = *reinterpret_cast<const v8bf*>(row32 + k0 + 8);
  v16bf a;
#pragma unroll
  for (int i = 0; i < 8; ++i) { a[i] = lo[i]; a[8 + i] = hi[i]; }
  return a;
}

// B-fragment (32x16, K x N): lane L holds column N = L&15; lanes 0-15 carry
// K 0..15, lanes 16-31 carry K 16..31 (16 contiguous bf16). `p` points at the
// lane's 16-element contiguous K run.
__device__ __forceinline__ v16bf ld_frag_b(const bf16_t* p) {
  v8bf lo = *reinterpret_cast<const v8bf*>(p);
  v8bf hi = *reinterpret_cast<const v8bf*>(p + 8);
  v16bf b;
#pragma unroll
  for (int i = 0; i < 8; ++i) { b[i] = lo[i]; b[8 + i] = hi[i]; }
  return b;
}

__device__ __forceinline__ void st4_bf16(bf16_t* dst, float4 v) {
  dst[0] = (bf16_t)v.x; dst[1] = (bf16_t)v.y;
  dst[2] = (bf16_t)v.z; dst[3] = (bf16_t)v.w;
}

// ---------------------------------------------------------------------------
// Kernel 1: fused QKV projection + RoPE.
// grid = (8192/128, 6144/128) ; block = 256 (8 waves).
// Block tile: 128 rows (x tokens) x 128 cols (one head of q/k/v).
// Wave: 16 rows x 128 cols = 8 WMMA accumulators => rotate-half pairing is
// acc[i] <-> acc[i+4] in-register.
// ---------------------------------------------------------------------------
__global__ __launch_bounds__(256)
void qkv_rope_kernel(const float* __restrict__ x,
                     const float* __restrict__ wq,
                     const float* __restrict__ wk,
                     const float* __restrict__ wv,
                     float* __restrict__ q_ws,
                     float* __restrict__ k_ws,
                     float* __restrict__ v_ws) {
  __shared__ __align__(16) bf16_t As[128][32];  // x tile      (8 KB)
  __shared__ __align__(16) bf16_t Bs[128][32];  // weight tile (8 KB)

  const int tid   = threadIdx.x;
  const int lane  = tid & 31;
  const int wid   = tid >> 5;
  const int nlane = lane & 15;
  const int mhalf = (lane < 16) ? 0 : 8;

  const int Mbase = blockIdx.x * 128;          // token-row base
  const int Nbase = blockIdx.y * 128;          // column in [0, 3*DIM)
  const int wmat  = Nbase >> 11;               // 0=q, 1=k, 2=v
  const int ncol0 = Nbase & (DIM - 1);         // column inside that projection
  const float* W  = (wmat == 0) ? wq : (wmat == 1 ? wk : wv);

  v8f acc[8];
#pragma unroll
  for (int i = 0; i < 8; ++i) acc[i] = (v8f){0.f,0.f,0.f,0.f,0.f,0.f,0.f,0.f};

  for (int kk = 0; kk < DIM / 32; ++kk) {
    const int k0 = kk * 32;
    // Stage A (x) and B (W) as bf16: 4 float4 loads each per thread.
#pragma unroll
    for (int u = 0; u < 4; ++u) {
      const int f   = tid * 4 + u;   // 0..1023 float4 slots
      const int row = f >> 3;        // 0..127
      const int c4  = f & 7;         // 0..7
      float4 av = *reinterpret_cast<const float4*>(
          x + (size_t)(Mbase + row) * DIM + k0 + c4 * 4);
      st4_bf16(&As[row][c4 * 4], av);
      float4 bv = *reinterpret_cast<const float4*>(
          W + (size_t)(ncol0 + row) * DIM + k0 + c4 * 4);
      st4_bf16(&Bs[row][c4 * 4], bv);
    }
    __syncthreads();

    if (kk + 1 < DIM / 32)  // pull next x slab toward L2 (global_prefetch_b8)
      __builtin_prefetch(x + (size_t)(Mbase + (tid >> 1)) * DIM + k0 + 32, 0, 1);

    v16bf a = ld_frag_a(&As[wid * 16 + nlane][0], lane);
#pragma unroll
    for (int i = 0; i < 8; ++i) {
      v16bf b = ld_frag_b(&Bs[i * 16 + nlane][(lane < 16) ? 0 : 16]);
      acc[i] = wmma_bf16(a, b, acc[i]);
    }
    __syncthreads();
  }

  // RoPE (q,k only) + scatter to [B, H, T, D] workspace.
  const int head = ncol0 >> 7;  // whole block lives in one head
  float invfreq[4];
  if (wmat < 2) {
#pragma unroll
    for (int i = 0; i < 4; ++i) {
      const float dmod = (float)((i * 16 + nlane) & 63);
      invfreq[i] = __expf(-(dmod * (1.0f / 64.0f)) * 9.21034037198f); // 10000^-d/64
    }
  }
  float* dst = (wmat == 0) ? q_ws : (wmat == 1 ? k_ws : v_ws);

#pragma unroll
  for (int r = 0; r < 8; ++r) {
    const int gm = Mbase + wid * 16 + mhalf + r;
    const int b  = gm >> 8;        // batch
    const int t  = gm & (SEQ - 1); // token
    float vals[8];
#pragma unroll
    for (int i = 0; i < 8; ++i) vals[i] = acc[i][r];
    if (wmat < 2) {
      const float tf = (float)t;
#pragma unroll
      for (int i = 0; i < 4; ++i) {
        float s, c;
        __sincosf(tf * invfreq[i], &s, &c);
        const float lo = vals[i], hi = vals[i + 4];
        vals[i]     = lo * c - hi * s;   // d < 64
        vals[i + 4] = hi * c + lo * s;   // d >= 64
      }
    }
    const size_t base = (((size_t)b * NH + head) * SEQ + t) * HD;
#pragma unroll
    for (int i = 0; i < 8; ++i) dst[base + i * 16 + nlane] = vals[i];
  }
}

// ---------------------------------------------------------------------------
// Kernel 2: causal attention for one (b, h, 64-query chunk).
// grid = B*NH*4 = 2048 ; block = 128 (4 waves, 16 query rows each).
// S = QK^T (WMMA), mask+softmax in-register (shfl_xor row reductions),
// P -> LDS (bf16, A-fragment order), O = P V (WMMA).
// ---------------------------------------------------------------------------
__global__ __launch_bounds__(128)
void attn_kernel(const float* __restrict__ q_ws,
                 const float* __restrict__ k_ws,
                 const float* __restrict__ v_ws,
                 float* __restrict__ o_ws) {
  __shared__ __align__(16) bf16_t KVs[256][32];       // K slab / V^T slab (16 KB)
  __shared__ __align__(16) bf16_t Pbuf[4][16][256];   // per-wave P        (32 KB)

  const int tid   = threadIdx.x;
  const int lane  = tid & 31;
  const int wid   = tid >> 5;
  const int nlane = lane & 15;
  const int mhalf = (lane < 16) ? 0 : 8;

  const int bh     = blockIdx.x >> 2;          // b*NH + h
  const int qchunk = blockIdx.x & 3;
  const int b      = bh >> 4;
  const int h      = bh & 15;
  const int qbase  = qchunk * 64 + wid * 16;   // wave's first query token
  const size_t kvb = (size_t)bh * SEQ * HD;

  v8f accs[16];
#pragma unroll
  for (int j = 0; j < 16; ++j) accs[j] = (v8f){0.f,0.f,0.f,0.f,0.f,0.f,0.f,0.f};

  // ---- S = Q K^T, looping 128-deep d in 4 chunks of 32 ----
  for (int dk = 0; dk < 4; ++dk) {
    // Stage K[256 keys][32 d] as bf16: 16 float4 per thread.
#pragma unroll
    for (int u = 0; u < 16; ++u) {
      const int f   = tid + u * 128;  // 0..2047 float4 slots
      const int key = f >> 3;
      const int c4  = f & 7;
      float4 kv = *reinterpret_cast<const float4*>(
          k_ws + kvb + (size_t)key * HD + dk * 32 + c4 * 4);
      st4_bf16(&KVs[key][c4 * 4], kv);
    }
    __syncthreads();

    // Q fragment straight from global (row-contiguous along d).
    const float* qp = q_ws + kvb + (size_t)(qbase + nlane) * HD + dk * 32 +
                      ((lane < 16) ? 0 : 16);
    float4 q0 = *reinterpret_cast<const float4*>(qp);
    float4 q1 = *reinterpret_cast<const float4*>(qp + 4);
    float4 q2 = *reinterpret_cast<const float4*>(qp + 8);
    float4 q3 = *reinterpret_cast<const float4*>(qp + 12);
    v16bf a;
    a[0]=(bf16_t)q0.x; a[1]=(bf16_t)q0.y; a[2]=(bf16_t)q0.z; a[3]=(bf16_t)q0.w;
    a[4]=(bf16_t)q1.x; a[5]=(bf16_t)q1.y; a[6]=(bf16_t)q1.z; a[7]=(bf16_t)q1.w;
    a[8]=(bf16_t)q2.x; a[9]=(bf16_t)q2.y; a[10]=(bf16_t)q2.z; a[11]=(bf16_t)q2.w;
    a[12]=(bf16_t)q3.x; a[13]=(bf16_t)q3.y; a[14]=(bf16_t)q3.z; a[15]=(bf16_t)q3.w;

#pragma unroll
    for (int j = 0; j < 16; ++j) {
      v16bf bf = ld_frag_b(&KVs[j * 16 + nlane][(lane < 16) ? 0 : 16]);
      accs[j] = wmma_bf16(a, bf, accs[j]);
    }
    __syncthreads();
  }

  // ---- causal mask + softmax (rows live at vgpr r, cols at lane&15) ----
  const float scale = 0.08838834764831845f;  // 1/sqrt(128)
  const float NEG = -__builtin_inff();
#pragma unroll
  for (int r = 0; r < 8; ++r) {
    const int qtok = qbase + mhalf + r;
    float m = NEG;
#pragma unroll
    for (int j = 0; j < 16; ++j) {
      const int ktok = j * 16 + nlane;
      const float s = (ktok <= qtok) ? accs[j][r] * scale : NEG;
      accs[j][r] = s;
      m = fmaxf(m, s);
    }
#pragma unroll
    for (int off = 1; off < 16; off <<= 1) m = fmaxf(m, __shfl_xor(m, off, 32));
    float sum = 0.f;
#pragma unroll
    for (int j = 0; j < 16; ++j) {
      const float p = __expf(accs[j][r] - m);
      accs[j][r] = p;
      sum += p;
    }
#pragma unroll
    for (int off = 1; off < 16; off <<= 1) sum += __shfl_xor(sum, off, 32);
    const float rinv = 1.0f / sum;
    const int mrow = mhalf + r;
#pragma unroll
    for (int j = 0; j < 16; ++j)
      Pbuf[wid][mrow][j * 16 + nlane] = (bf16_t)(accs[j][r] * rinv);
  }

  // ---- O = P V, looping 256 keys in 8 chunks of 32 ----
  v8f acco[8];
#pragma unroll
  for (int i = 0; i < 8; ++i) acco[i] = (v8f){0.f,0.f,0.f,0.f,0.f,0.f,0.f,0.f};

  for (int kc = 0; kc < 8; ++kc) {
    __syncthreads();  // everyone done reading previous KVs contents
    // Stage V^T: KVs[d][key] (d 0..127, key 0..31): 8 float4 per thread.
#pragma unroll
    for (int u = 0; u < 8; ++u) {
      const int f   = tid + u * 128;  // 0..1023 float4 slots over V[32][128]
      const int key = f >> 5;
      const int c4  = f & 31;         // d/4
      float4 vv = *reinterpret_cast<const float4*>(
          v_ws + kvb + (size_t)(kc * 32 + key) * HD + c4 * 4);
      KVs[c4 * 4 + 0][key] = (bf16_t)vv.x;
      KVs[c4 * 4 + 1][key] = (bf16_t)vv.y;
      KVs[c4 * 4 + 2][key] = (bf16_t)vv.z;
      KVs[c4 * 4 + 3][key] = (bf16_t)vv.w;
    }
    __syncthreads();

    v16bf a = ld_frag_a(&Pbuf[wid][nlane][kc * 32], lane);
#pragma unroll
    for (int i = 0; i < 8; ++i) {
      v16bf bf = ld_frag_b(&KVs[i * 16 + nlane][(lane < 16) ? 0 : 16]);
      acco[i] = wmma_bf16(a, bf, acco[i]);
    }
  }

  // Store O as [B, T, H, D] (== [8192, 2048] rows for the final GEMM).
#pragma unroll
  for (int r = 0; r < 8; ++r) {
    const int t = qbase + mhalf + r;
    const size_t base = (((size_t)b * SEQ + t) * NH + h) * HD;
#pragma unroll
    for (int i = 0; i < 8; ++i) o_ws[base + i * 16 + nlane] = acco[i][r];
  }
}

// ---------------------------------------------------------------------------
// Kernel 3: output projection out = O @ Wo^T.
// grid = (8192/128, 2048/128) ; block = 256 (8 waves).
// ---------------------------------------------------------------------------
__global__ __launch_bounds__(256)
void out_proj_kernel(const float* __restrict__ o_ws,
                     const float* __restrict__ wo,
                     float* __restrict__ out) {
  __shared__ __align__(16) bf16_t As[128][32];
  __shared__ __align__(16) bf16_t Bs[128][32];

  const int tid   = threadIdx.x;
  const int lane  = tid & 31;
  const int wid   = tid >> 5;
  const int nlane = lane & 15;
  const int mhalf = (lane < 16) ? 0 : 8;
  const int Mbase = blockIdx.x * 128;
  const int Nbase = blockIdx.y * 128;

  v8f acc[8];
#pragma unroll
  for (int i = 0; i < 8; ++i) acc[i] = (v8f){0.f,0.f,0.f,0.f,0.f,0.f,0.f,0.f};

  for (int kk = 0; kk < DIM / 32; ++kk) {
    const int k0 = kk * 32;
#pragma unroll
    for (int u = 0; u < 4; ++u) {
      const int f   = tid * 4 + u;
      const int row = f >> 3;
      const int c4  = f & 7;
      float4 av = *reinterpret_cast<const float4*>(
          o_ws + (size_t)(Mbase + row) * DIM + k0 + c4 * 4);
      st4_bf16(&As[row][c4 * 4], av);
      float4 bv = *reinterpret_cast<const float4*>(
          wo + (size_t)(Nbase + row) * DIM + k0 + c4 * 4);
      st4_bf16(&Bs[row][c4 * 4], bv);
    }
    __syncthreads();
    v16bf a = ld_frag_a(&As[wid * 16 + nlane][0], lane);
#pragma unroll
    for (int i = 0; i < 8; ++i) {
      v16bf b = ld_frag_b(&Bs[i * 16 + nlane][(lane < 16) ? 0 : 16]);
      acc[i] = wmma_bf16(a, b, acc[i]);
    }
    __syncthreads();
  }

#pragma unroll
  for (int r = 0; r < 8; ++r) {
    const int gm = Mbase + wid * 16 + mhalf + r;
#pragma unroll
    for (int i = 0; i < 8; ++i)
      out[(size_t)gm * DIM + Nbase + i * 16 + nlane] = acc[i][r];
  }
}

// ---------------------------------------------------------------------------
extern "C" void kernel_launch(void* const* d_in, const int* in_sizes, int n_in,
                              void* d_out, int out_size, void* d_ws, size_t ws_size,
                              hipStream_t stream) {
  (void)in_sizes; (void)n_in; (void)out_size; (void)ws_size;
  const float* x  = (const float*)d_in[0];
  // d_in[1] = causal mask (recomputed analytically on device; unused)
  const float* wq = (const float*)d_in[2];
  const float* wk = (const float*)d_in[3];
  const float* wv = (const float*)d_in[4];
  const float* wo = (const float*)d_in[5];
  float* out = (float*)d_out;

  const size_t elems = (size_t)MROWS * DIM;  // 16.78M f32 = 64 MB
  float* q_ws = (float*)d_ws;
  float* k_ws = q_ws + elems;
  float* v_ws = k_ws + elems;
  float* o_ws = v_ws + elems;

  qkv_rope_kernel<<<dim3(MROWS / 128, (3 * DIM) / 128), 256, 0, stream>>>(
      x, wq, wk, wv, q_ws, k_ws, v_ws);
  attn_kernel<<<dim3(BATCH * NH * 4), 128, 0, stream>>>(q_ws, k_ws, v_ws, o_ws);
  out_proj_kernel<<<dim3(MROWS / 128, DIM / 128), 256, 0, stream>>>(o_ws, wo, out);
}
